// DetectorInferenceLayer_6992206758132
// MI455X (gfx1250) — compile-verified, matrix operand order checked by reading, not run on hardware
//
#include <hip/hip_runtime.h>
#include <stdint.h>

// ---------------- problem constants ----------------
#define GH 128
#define GW 128
#define NA 9
#define NANCH (GH*GW*NA)      // 147456
#define NC 86
#define NB 8
#define NBINS 16384           // 14-bit prefix of sortable score
#define BIN_SHIFT 18
#define TOPK 1000
#define CAND_CAP 2048
#define MAXDET 100
#define SCORE_TH 0.5f
#define IOU_TH 0.75f

// ---------------- workspace layout (bytes), ~1.94 MB total ----------------
#define OFF_HIST 0u                                   // NB*NBINS*4  = 524288
#define OFF_CNT  (OFF_HIST + NB*NBINS*4u)             // 8 u32 (zeroed with hist)
#define OFF_CUT  (OFF_CNT + 64u)
#define OFF_VAL  (OFF_CUT + 64u)
#define OFF_KEYS (OFF_VAL + 64u)                      // NB*2048*8   = 131072
#define OFF_CORN (OFF_KEYS + NB*CAND_CAP*8u)          // NB*1024*4*4 = 131072
#define OFF_OUT6 (OFF_CORN + NB*1024u*16u)            // NB*1024*6*4 = 196608
#define OFF_MASK (OFF_OUT6 + NB*1024u*24u)            // NB*1024*64*2= 1048576

typedef float    v8f  __attribute__((ext_vector_type(8)));
typedef float    v2f  __attribute__((ext_vector_type(2)));
typedef _Float16 v16h __attribute__((ext_vector_type(16)));
typedef int      v4i  __attribute__((vector_size(4 * sizeof(int))));   // matches builtin proto
typedef __attribute__((address_space(1))) v4i* gv4i_p;                 // global int4*
typedef __attribute__((address_space(3))) v4i* lv4i_p;                 // LDS int4*

__device__ __forceinline__ unsigned sortable_bits(float s) {
    unsigned u = __float_as_uint(s);
    return (u & 0x80000000u) ? ~u : (u | 0x80000000u);
}

// ---------------- 0: zero histograms + per-image counters ----------------
__global__ void kInit(unsigned* p, int n) {
    int i = blockIdx.x * 256 + threadIdx.x;
    if (i < n) p[i] = 0u;
}

// ---------------- 1: radix histogram of score channel ----------------
__global__ __launch_bounds__(256) void kHist(const float* __restrict__ in, unsigned* __restrict__ hist) {
    int img = blockIdx.y;
    int idx = blockIdx.x * 256 + threadIdx.x;
    float s = in[((size_t)img * NANCH + idx) * NC + 5];
    if (s >= SCORE_TH) {
        unsigned key = sortable_bits(s);
        atomicAdd(&hist[(size_t)img * NBINS + (key >> BIN_SHIFT)], 1u);
    }
}

// ---------------- 2: find top-K cutoff bin per image ----------------
__global__ __launch_bounds__(256) void kCutoff(const unsigned* __restrict__ hist, unsigned* __restrict__ cutoff) {
    __shared__ unsigned part[256];
    __shared__ unsigned pref[256];
    int img = blockIdx.x, t = threadIdx.x;
    const unsigned* h = hist + (size_t)img * NBINS;
    int hi = NBINS - 1 - t * 64;              // chunk t covers bins hi .. hi-63 (descending)
    unsigned sum = 0;
    for (int b = 0; b < 64; ++b) sum += h[hi - b];
    part[t] = sum;
    __syncthreads();
    if (t == 0) {
        unsigned run = 0;
        for (int i = 0; i < 256; ++i) { pref[i] = run; run += part[i]; }
        if (run < TOPK) cutoff[img] = 0u;     // fewer than TOPK pass threshold: take all
    }
    __syncthreads();
    if (pref[t] < TOPK && pref[t] + part[t] >= TOPK) {   // exactly one chunk crosses
        unsigned cum = pref[t];
        for (int b = 0; b < 64; ++b) {
            cum += h[hi - b];
            if (cum >= TOPK) {
                int cutBin = hi - b;
                if (cum > CAND_CAP && cutBin < NBINS - 1) { cum -= h[cutBin]; cutBin += 1; }
                cutoff[img] = ((unsigned)cutBin) << BIN_SHIFT;
                break;
            }
        }
    }
}

// ---------------- 3: compact candidates (L2-hot re-read) ----------------
__global__ __launch_bounds__(256) void kCompact(const float* __restrict__ in, const unsigned* __restrict__ cutoff,
                                                unsigned* __restrict__ cnt, unsigned long long* __restrict__ keys) {
    int img = blockIdx.y;
    unsigned idx = blockIdx.x * 256 + threadIdx.x;
    float s = in[((size_t)img * NANCH + idx) * NC + 5];
    if (s >= SCORE_TH) {
        unsigned key = sortable_bits(s);
        if (key >= cutoff[img]) {
            unsigned pos = atomicAdd(&cnt[img], 1u);
            if (pos < CAND_CAP)   // key low32 = ~idx -> total order, deterministic after sort
                keys[(size_t)img * CAND_CAP + pos] =
                    ((unsigned long long)key << 32) | (unsigned)(~idx);
        }
    }
}

// ---------------- 4: sort, decode anchors, argmax label ----------------
__global__ __launch_bounds__(1024) void kPrep(const float* __restrict__ in,
                                              const unsigned long long* __restrict__ candKeys,
                                              const unsigned* __restrict__ candCnt,
                                              unsigned* __restrict__ validCnt,
                                              float* __restrict__ corners, float* __restrict__ out6) {
    __shared__ unsigned long long keys[CAND_CAP];
    __shared__ float ldsBox[1024 * 4];
    int img = blockIdx.x, tid = threadIdx.x;
    unsigned cnt = candCnt[img];
    if (cnt > CAND_CAP) cnt = CAND_CAP;
    for (int i = tid; i < CAND_CAP; i += 1024)
        keys[i] = (i < (int)cnt) ? candKeys[(size_t)img * CAND_CAP + i] : 0ull;
    __syncthreads();
    // bitonic sort, descending (valid keys have MSB set, zeros sink to the end)
    for (unsigned k = 2; k <= CAND_CAP; k <<= 1)
        for (unsigned j = k >> 1; j > 0; j >>= 1) {
            __syncthreads();
            unsigned t = tid;
            unsigned i = 2u * t - (t & (j - 1u));
            unsigned p = i + j;
            bool asc = ((i & k) == 0u);
            unsigned long long a = keys[i], b = keys[p];
            bool sw = asc ? (a < b) : (a > b);
            if (sw) { keys[i] = b; keys[p] = a; }
        }
    __syncthreads();
    int V = (int)cnt; if (V > TOPK) V = TOPK;
    if (tid == 0) validCnt[img] = (unsigned)V;

    float o0=0,o1=0,o2=0,o3=0,o4=0,o5=0, c0=0,c1=0,c2=0,c3=0;
    if (tid < V) {
        unsigned long long key = keys[tid];
        unsigned idx = ~(unsigned)(key & 0xFFFFFFFFull);
        int a = (int)(idx % NA); int cell = (int)(idx / NA);
        int gj = cell % GW, gi = cell / GW;
        const float sTab[3] = {0.5f, 1.0f, 2.0f};
        float s = sTab[a / 3], r = sTab[a % 3];
        float sr = sqrtf(r);
        float ah = s * sr * (4.0f / GH), aw = (s / sr) * (4.0f / GW);
        float ci = (gi + 0.5f) / GH, cj = (gj + 0.5f) / GW;
        const float* g = in + ((size_t)img * NANCH + idx) * NC;
        float t0, t1, t2, t3;
#if __has_builtin(__builtin_amdgcn_global_load_async_to_lds_b128)
        // CDNA5 async copy: stage the 16B box-offset vector through LDS (ASYNCcnt path)
        __builtin_amdgcn_global_load_async_to_lds_b128(
            (gv4i_p)g, (lv4i_p)&ldsBox[tid * 4], 0, 0);
    #if __has_builtin(__builtin_amdgcn_s_wait_asynccnt)
        __builtin_amdgcn_s_wait_asynccnt(0);
    #else
        asm volatile("s_wait_asynccnt 0" ::: "memory");
    #endif
        t0 = ldsBox[tid*4+0]; t1 = ldsBox[tid*4+1]; t2 = ldsBox[tid*4+2]; t3 = ldsBox[tid*4+3];
#else
        t0 = g[0]; t1 = g[1]; t2 = g[2]; t3 = g[3];
#endif
        float score = g[5];
        __builtin_prefetch(g + 6, 0, 3);            // global_prefetch_b8 ahead of logit scan
        float best = g[6]; int bl = 0;
        for (int cc = 1; cc < 80; ++cc) { float v = g[6 + cc]; if (v > best) { best = v; bl = cc; } }
        float di = ci + t0 * ah, dj = cj + t1 * aw;
        float dh = ah * __expf(t2), dw = aw * __expf(t3);
        o0 = di; o1 = dj; o2 = dh; o3 = dw; o4 = (float)bl; o5 = score;
        c0 = di - dh * 0.5f; c1 = dj - dw * 0.5f; c2 = di + dh * 0.5f; c3 = dj + dw * 0.5f;
    }
    float* cr = corners + ((size_t)img * 1024 + tid) * 4;
    cr[0] = c0; cr[1] = c1; cr[2] = c2; cr[3] = c3;
    float* od = out6 + ((size_t)img * 1024 + tid) * 6;
    od[0] = o0; od[1] = o1; od[2] = o2; od[3] = o3; od[4] = o4; od[5] = o5;
}

// ---------------- 5: parallel tiled suppression mask (one wave32 per 16x16 tile) -----
// union base (area_i + area_j) for the tile computed as a rank-2 update on the
// matrix pipe via v_wmma_f32_16x16x4_f32; intersection min/max co-executes on VALU.
// IoU test is division-free: inter > IOU_TH * union  (with inter > 0 guard).
__global__ __launch_bounds__(32) void kMask(const float* __restrict__ corners,
                                            const unsigned* __restrict__ validCnt,
                                            unsigned short* __restrict__ mask) {
    int lane = threadIdx.x;
    int tc = blockIdx.x, tr = blockIdx.y, img = blockIdx.z;
    int V = (int)validCnt[img];
    int hl = lane >> 4, lm = lane & 15;
    int ncol = tc * 16 + lm;                    // column box owned by this lane
    const float* cb = corners + ((size_t)img * 1024 + ncol) * 4;
    float cti = cb[0], ctj = cb[1], cbi = cb[2], cbj = cb[3];
    float areaC = (cbi - cti) * (cbj - ctj);
    const float* ra = corners + ((size_t)img * 1024 + tr * 16 + lm) * 4;
    float areaRa = (ra[2] - ra[0]) * (ra[3] - ra[1]);
    v8f d;
#if __has_builtin(__builtin_amdgcn_wmma_f32_16x16x4_f32)
    {
        // A(16x4): col0 = row areas, col1 = ones.  B(4x16): row0 = ones, row1 = col areas.
        v2f a, b;
        a.x = hl ? 0.f : areaRa;  a.y = hl ? 0.f : 1.0f;   // lanes 0-15: K0,K1 ; lanes 16-31: K2,K3 = 0
        b.x = hl ? 0.f : 1.0f;    b.y = hl ? 0.f : areaC;
        v8f c = {0.f,0.f,0.f,0.f,0.f,0.f,0.f,0.f};
        d = __builtin_amdgcn_wmma_f32_16x16x4_f32(false, a, false, b, (short)0, c, false, false);
    }
#else
    {
        v16h a, b;
        for (int q = 0; q < 16; ++q) { a[q] = (_Float16)0.f; b[q] = (_Float16)0.f; }
        if (!hl) { a[0] = (_Float16)areaRa; a[1] = (_Float16)1.0f;
                   b[0] = (_Float16)1.0f;  b[1] = (_Float16)areaC; }
        v8f c = {0.f,0.f,0.f,0.f,0.f,0.f,0.f,0.f};
        d = __builtin_amdgcn_wmma_f32_16x16x32_f16(false, a, false, b, (short)0, c, false, false);
    }
#endif
    for (int v = 0; v < 8; ++v) {
        int m = tr * 16 + v + 8 * hl;           // D row held by this lane for VGPR v
        const float* rb = corners + ((size_t)img * 1024 + m) * 4;
        float ti = fmaxf(rb[0], cti), tj = fmaxf(rb[1], ctj);
        float bi = fminf(rb[2], cbi), bj = fminf(rb[3], cbj);
        float inter = fmaxf(bi - ti, 0.f) * fmaxf(bj - tj, 0.f);
        float uni = d[v] - inter;
        // iou > TH  <=>  inter > TH * union (union >= inter >= 0 whenever it matters)
        bool pred = (inter > 0.f) & (inter > IOU_TH * uni)
                  & (ncol > m) & (ncol < V) & (m < V);
        unsigned bal;
#if __has_builtin(__builtin_amdgcn_ballot_w32)
        bal = __builtin_amdgcn_ballot_w32(pred);
#else
        bal = (unsigned)__ballot(pred);
#endif
        if (lane == 0) {
            mask[((size_t)img * 1024 + tr * 16 + v) * 64 + tc]     = (unsigned short)(bal & 0xFFFFu);
            mask[((size_t)img * 1024 + tr * 16 + v + 8) * 64 + tc] = (unsigned short)(bal >> 16);
        }
    }
}

// ---------------- 6: serial bitmask-OR greedy reduce + emit ----------------
__global__ __launch_bounds__(128) void kFinal(const unsigned short* __restrict__ mask,
                                              const unsigned* __restrict__ validCnt,
                                              const float* __restrict__ out6, float* __restrict__ out) {
    __shared__ unsigned char sup[1024];
    __shared__ int keptList[MAXDET];
    __shared__ int keptCount;
    __shared__ int cur;
    int img = blockIdx.x, t = threadIdx.x;
    int V = (int)validCnt[img];
    for (int i = t; i < 1024; i += 128) sup[i] = 0;
    if (t == 0) keptCount = 0;
    __syncthreads();
    for (int i = 0; i < V; ++i) {
        if (t == 0) {
            int k = (sup[i] == 0);
            cur = k;
            if (k) {
                keptList[keptCount] = i;
                keptCount++;
                if (keptCount >= MAXDET) cur = 2;   // 100th keep recorded: stop
            }
        }
        __syncthreads();
        if (cur == 2) break;
        if (cur == 1 && t < 64) {
            unsigned m = mask[((size_t)img * 1024 + i) * 64 + t];
            if (m) for (int b = 0; b < 16; ++b) if ((m >> b) & 1u) sup[t * 16 + b] = 1;
        }
        __syncthreads();
    }
    __syncthreads();
    int kc = keptCount; if (kc > MAXDET) kc = MAXDET;
    for (int k = t; k < MAXDET; k += 128) {
        const float* src = (k < kc) ? (out6 + ((size_t)img * 1024 + keptList[k]) * 6) : (const float*)0;
        for (int c2 = 0; c2 < 6; ++c2)
            out[(size_t)img * (MAXDET * 6) + k * 6 + c2] = src ? src[c2] : 0.0f;
    }
}

// ---------------- host launcher ----------------
extern "C" void kernel_launch(void* const* d_in, const int* in_sizes, int n_in,
                              void* d_out, int out_size, void* d_ws, size_t ws_size,
                              hipStream_t stream) {
    (void)in_sizes; (void)n_in; (void)out_size; (void)ws_size;
    const float* in = (const float*)d_in[0];
    char* ws = (char*)d_ws;
    unsigned* hist = (unsigned*)(ws + OFF_HIST);
    unsigned* cnt  = (unsigned*)(ws + OFF_CNT);
    unsigned* cut  = (unsigned*)(ws + OFF_CUT);
    unsigned* val  = (unsigned*)(ws + OFF_VAL);
    unsigned long long* keys = (unsigned long long*)(ws + OFF_KEYS);
    float* corners = (float*)(ws + OFF_CORN);
    float* out6    = (float*)(ws + OFF_OUT6);
    unsigned short* mask = (unsigned short*)(ws + OFF_MASK);
    float* out = (float*)d_out;

    int nz = NB * NBINS + 8;                       // hist + contiguous counters
    kInit<<<(nz + 255) / 256, 256, 0, stream>>>(hist, nz);
    kHist<<<dim3(NANCH / 256, NB), 256, 0, stream>>>(in, hist);
    kCutoff<<<NB, 256, 0, stream>>>(hist, cut);
    kCompact<<<dim3(NANCH / 256, NB), 256, 0, stream>>>(in, cut, cnt, keys);
    kPrep<<<NB, 1024, 0, stream>>>(in, keys, cnt, val, corners, out6);
    kMask<<<dim3(64, 64, NB), 32, 0, stream>>>(corners, val, mask);
    kFinal<<<NB, 128, 0, stream>>>(mask, val, out6, out);
}
